// Head_89000312307936
// MI455X (gfx1250) — matrix operand order repaired and checked by reading, not compiled, not run on hardware
//
#include <hip/hip_runtime.h>

typedef _Float16 half_t;
typedef __attribute__((ext_vector_type(16))) _Float16 v16h;
typedef __attribute__((ext_vector_type(8)))  _Float16 h8;
typedef __attribute__((ext_vector_type(2)))  _Float16 h2;
typedef __attribute__((ext_vector_type(8)))  float    v8f;
typedef __attribute__((ext_vector_type(4)))  float    f32x4;
typedef __attribute__((ext_vector_type(8)))  unsigned u32x8;

#define TSEQ 256
#define CDIM 1024
#define HDIM 64

static __device__ inline v8f vzero8() {
    v8f z;
#pragma unroll
    for (int i = 0; i < 8; ++i) z[i] = 0.0f;
    return z;
}

// Assemble a 16x32-f16 WMMA A-operand lane fragment from two contiguous 16B chunks:
// lane holds K = {k0..k0+15} contiguous (k0 = 16*half).
static __device__ inline v16h load_a16(const half_t* __restrict__ p) {
    h8 lo = *(const h8*)p;
    h8 hi = *(const h8*)(p + 8);
    v16h a;
#pragma unroll
    for (int i = 0; i < 8; ++i) { a[i] = lo[i]; a[i + 8] = hi[i]; }
    return a;
}

static __device__ inline unsigned pkh2(float a, float b) {
    h2 t;
    t[0] = (half_t)a;
    t[1] = (half_t)b;
    return __builtin_bit_cast(unsigned, t);
}

// ---------------------------------------------------------------------------
// Kernel 0: transpose + fp32->f16 convert the three weight matrices.
// Wt[h*1024 + c] = (f16) W[c*64 + h]
// ---------------------------------------------------------------------------
__global__ __launch_bounds__(256) void wt_kernel(
    const float* __restrict__ Wq, const float* __restrict__ Wk,
    const float* __restrict__ Wv, half_t* __restrict__ wtq,
    half_t* __restrict__ wtk, half_t* __restrict__ wtv) {
    int idx = blockIdx.x * 256 + threadIdx.x;  // 0..65535
    int h = idx >> 10;
    int c = idx & 1023;
    wtq[idx] = (half_t)Wq[c * HDIM + h];
    wtk[idx] = (half_t)Wk[c * HDIM + h];
    wtv[idx] = (half_t)Wv[c * HDIM + h];
}

// ---------------------------------------------------------------------------
// Kernel 1: QKV projection.  One wave computes a 16x64 tile of Q, K, and V.
// A = x rows (fp32 -> f16 in-register), B = Wt rows (contiguous f16).
// Q scaled by 1/sqrt(H); V stored transposed per batch: vt[b][h][t].
// ---------------------------------------------------------------------------
__global__ __launch_bounds__(256) void qkv_kernel(
    const float* __restrict__ x,
    const half_t* __restrict__ wtq, const half_t* __restrict__ wtk,
    const half_t* __restrict__ wtv,
    const float* __restrict__ bq, const float* __restrict__ bk,
    const float* __restrict__ bv,
    half_t* __restrict__ qh, half_t* __restrict__ kh,
    half_t* __restrict__ vt) {
    const int lane = threadIdx.x & 31;
    const int wave = threadIdx.x >> 5;
    const int tile = blockIdx.x * 8 + wave;   // 0..4095
    const int m0 = tile * 16;                 // row base in flattened [65536, C]
    const int n = lane & 15;
    const int hf = lane >> 4;

    v8f acc[12];
#pragma unroll
    for (int i = 0; i < 12; ++i) acc[i] = vzero8();

    const float* xrow = x + (size_t)(m0 + n) * CDIM;
    const half_t* wt0 = wtq + (size_t)n * CDIM;
    const half_t* wt1 = wtk + (size_t)n * CDIM;
    const half_t* wt2 = wtv + (size_t)n * CDIM;

    for (int kk = 0; kk < CDIM; kk += 32) {
        // A fragment: x[m0+n][kk + 16*hf + {0..15}]
        const float* xp = xrow + kk + hf * 16;
        f32x4 a0 = *(const f32x4*)(xp);
        f32x4 a1 = *(const f32x4*)(xp + 4);
        f32x4 a2 = *(const f32x4*)(xp + 8);
        f32x4 a3 = *(const f32x4*)(xp + 12);
        v16h a;
#pragma unroll
        for (int i = 0; i < 4; ++i) {
            a[i]      = (half_t)a0[i];
            a[4 + i]  = (half_t)a1[i];
            a[8 + i]  = (half_t)a2[i];
            a[12 + i] = (half_t)a3[i];
        }
        const int boff = kk + hf * 16;
#pragma unroll
        for (int t = 0; t < 4; ++t) {
            v16h b0 = *(const v16h*)(wt0 + (size_t)t * 16 * CDIM + boff);
            acc[t] = __builtin_amdgcn_wmma_f32_16x16x32_f16(
                false, a, false, b0, (short)0, acc[t], false, false);
            v16h b1 = *(const v16h*)(wt1 + (size_t)t * 16 * CDIM + boff);
            acc[4 + t] = __builtin_amdgcn_wmma_f32_16x16x32_f16(
                false, a, false, b1, (short)0, acc[4 + t], false, false);
            v16h b2 = *(const v16h*)(wt2 + (size_t)t * 16 * CDIM + boff);
            acc[8 + t] = __builtin_amdgcn_wmma_f32_16x16x32_f16(
                false, a, false, b2, (short)0, acc[8 + t], false, false);
        }
    }

    // Epilogue: bias, Q scale, stores.  C/D layout: elem(M=r+8*hf, N=n).
#pragma unroll
    for (int t = 0; t < 4; ++t) {
        const int col = t * 16 + n;
        const float bqv = bq[col], bkv = bk[col], bvv = bv[col];
#pragma unroll
        for (int r = 0; r < 8; ++r) {
            const int R = m0 + r + 8 * hf;  // global flattened row
            qh[(size_t)R * HDIM + col] = (half_t)((acc[t][r] + bqv) * 0.125f);
            kh[(size_t)R * HDIM + col] = (half_t)(acc[4 + t][r] + bkv);
            const int b = R >> 8, tp = R & 255;
            vt[((size_t)b * HDIM + col) * TSEQ + tp] =
                (half_t)(acc[8 + t][r] + bvv);
        }
    }
}

// ---------------------------------------------------------------------------
// Kernel 2: flash attention.  One wave = (batch, 16-query tile).
// St = K_blk(32x64) @ Q^T(64x16)  -> softmax state fully per-lane (query=N dim)
// O^T(64x16) += V^T(64x32) @ P^T(32x16)
// ---------------------------------------------------------------------------
__global__ __launch_bounds__(256) void attn_kernel(
    const half_t* __restrict__ qh, const half_t* __restrict__ kh,
    const half_t* __restrict__ vt, float* __restrict__ out) {
    const int lane = threadIdx.x & 31;
    const int wave = threadIdx.x >> 5;
    const int gw = blockIdx.x * 8 + wave;  // 0..4095
    const int batch = gw >> 4;
    const int m0 = (gw & 15) * 16;
    const int n = lane & 15;
    const int hf = lane >> 4;

    const half_t* Q = qh + (size_t)batch * TSEQ * HDIM;
    const half_t* K = kh + (size_t)batch * TSEQ * HDIM;
    const half_t* V = vt + (size_t)batch * HDIM * TSEQ;

    // Q as B-operand (column n of Q^T = Q row m0+n, contiguous), 2 head-chunks
    v16h qb0 = *(const v16h*)(Q + (size_t)(m0 + n) * HDIM + hf * 16);
    v16h qb1 = *(const v16h*)(Q + (size_t)(m0 + n) * HDIM + 32 + hf * 16);

    float mi = -__builtin_inff();
    float li = 0.0f;
    v8f o[4];
#pragma unroll
    for (int i = 0; i < 4; ++i) o[i] = vzero8();

    const int nblocks = (m0 + 47) >> 5;  // causal key blocks of 32
    for (int j = 0; j < nblocks; ++j) {
        const int kb0 = j * 32;
        // ---- St = K_block @ Q^T : two 16-key M-tiles, K-dim = head(64) ----
        v8f st[2];
#pragma unroll
        for (int t = 0; t < 2; ++t) {
            const half_t* kr = K + (size_t)(kb0 + t * 16 + n) * HDIM + hf * 16;
            v8f s = vzero8();
            v16h a0 = load_a16(kr);
            s = __builtin_amdgcn_wmma_f32_16x16x32_f16(false, a0, false, qb0,
                                                       (short)0, s, false, false);
            v16h a1 = load_a16(kr + 32);
            s = __builtin_amdgcn_wmma_f32_16x16x32_f16(false, a1, false, qb1,
                                                       (short)0, s, false, false);
            st[t] = s;
        }
        // ---- causal mask: elem(key = kb0 + t*16 + r + 8*hf, query = m0+n) ----
#pragma unroll
        for (int t = 0; t < 2; ++t)
#pragma unroll
            for (int r = 0; r < 8; ++r) {
                const int key = kb0 + t * 16 + r + 8 * hf;
                if (key > m0 + n) st[t][r] = -__builtin_inff();
            }
        // ---- online softmax (per-lane query; combine lane halves via xor16) --
        float mx = st[0][0];
#pragma unroll
        for (int t = 0; t < 2; ++t)
#pragma unroll
            for (int r = 0; r < 8; ++r) mx = fmaxf(mx, st[t][r]);
        mx = fmaxf(mx, __shfl_xor(mx, 16));
        const float mnew = fmaxf(mi, mx);
        const float scale = __expf(mi - mnew);
        float rsum = 0.0f;
#pragma unroll
        for (int t = 0; t < 2; ++t)
#pragma unroll
            for (int r = 0; r < 8; ++r) {
                st[t][r] = __expf(st[t][r] - mnew);
                rsum += st[t][r];
            }
        rsum += __shfl_xor(rsum, 16);
        li = li * scale + rsum;
        mi = mnew;
#pragma unroll
        for (int i = 0; i < 4; ++i)
#pragma unroll
            for (int r = 0; r < 8; ++r) o[i][r] *= scale;

        // ---- pack P^T into B-operand: lane(q=n,half=hf) needs keys 16*hf+{0..15}
        unsigned pt0[4], pt1[4], ot0[4], ot1[4];
#pragma unroll
        for (int jj = 0; jj < 4; ++jj) {
            pt0[jj] = pkh2(st[0][2 * jj], st[0][2 * jj + 1]);  // keys 8*hf+{2j,2j+1}
            pt1[jj] = pkh2(st[1][2 * jj], st[1][2 * jj + 1]);  // keys 16+8*hf+...
        }
#pragma unroll
        for (int jj = 0; jj < 4; ++jj) {
            ot0[jj] = (unsigned)__shfl_xor((int)pt0[jj], 16);
            ot1[jj] = (unsigned)__shfl_xor((int)pt1[jj], 16);
        }
        u32x8 bb;
#pragma unroll
        for (int jj = 0; jj < 4; ++jj) {
            bb[jj]     = (hf == 0) ? pt0[jj] : ot1[jj];  // keys 16h+{0..7}
            bb[4 + jj] = (hf == 0) ? ot0[jj] : pt1[jj];  // keys 16h+{8..15}
        }
        v16h pb = __builtin_bit_cast(v16h, bb);

        // ---- O^T += V^T_tile(16col x 32key) @ P^T ----
#pragma unroll
        for (int ct = 0; ct < 4; ++ct) {
            const half_t* vr =
                V + (size_t)(ct * 16 + n) * TSEQ + kb0 + hf * 16;
            v16h a = load_a16(vr);
            o[ct] = __builtin_amdgcn_wmma_f32_16x16x32_f16(
                false, a, false, pb, (short)0, o[ct], false, false);
        }
    }

    // ---- normalize + store: O^T elem(col = ct*16 + r + 8*hf, query = n) ----
    const float inv = 1.0f / li;
    float* orow = out + ((size_t)batch * TSEQ + m0 + n) * HDIM;
#pragma unroll
    for (int ct = 0; ct < 4; ++ct)
#pragma unroll
        for (int r = 0; r < 8; ++r)
            orow[ct * 16 + r + 8 * hf] = o[ct][r] * inv;
}

// ---------------------------------------------------------------------------
// Workspace layout (bytes):
//   [0)        Wt_q  f16 64x1024   131072
//   [131072)   Wt_k                131072
//   [262144)   Wt_v                131072
//   [393216)   Qh    f16 65536x64  8388608   (pre-scaled by 1/8)
//   [+8388608) Kh    f16 65536x64  8388608
//   [+8388608) Vt    f16 256x64x256 8388608  (per-batch transposed V)
// total ~24.4 MB
// ---------------------------------------------------------------------------
extern "C" void kernel_launch(void* const* d_in, const int* in_sizes, int n_in,
                              void* d_out, int out_size, void* d_ws,
                              size_t ws_size, hipStream_t stream) {
    const float* x  = (const float*)d_in[0];
    const float* Wq = (const float*)d_in[1];
    const float* Wk = (const float*)d_in[2];
    const float* Wv = (const float*)d_in[3];
    const float* bq = (const float*)d_in[4];
    const float* bk = (const float*)d_in[5];
    const float* bv = (const float*)d_in[6];
    float* out = (float*)d_out;

    char* ws = (char*)d_ws;
    half_t* wtq = (half_t*)(ws);
    half_t* wtk = (half_t*)(ws + 131072);
    half_t* wtv = (half_t*)(ws + 262144);
    half_t* qhp = (half_t*)(ws + 393216);
    half_t* khp = (half_t*)(ws + 393216 + 8388608);
    half_t* vtp = (half_t*)(ws + 393216 + 2 * 8388608);

    wt_kernel<<<256, 256, 0, stream>>>(Wq, Wk, Wv, wtq, wtk, wtv);
    qkv_kernel<<<512, 256, 0, stream>>>(x, wtq, wtk, wtv, bq, bk, bv, qhp, khp,
                                        vtp);
    attn_kernel<<<512, 256, 0, stream>>>(qhp, khp, vtp, out);
}
